// LocalLayer_26156350832862
// MI455X (gfx1250) — matrix operand order repaired and checked by reading, not compiled, not run on hardware
//
#include <hip/hip_runtime.h>

typedef _Float16 v16h __attribute__((ext_vector_type(16)));
typedef _Float16 v8h  __attribute__((ext_vector_type(8)));
typedef float    v8f  __attribute__((ext_vector_type(8)));
typedef float    v4f  __attribute__((ext_vector_type(4)));

#define DGRID  16
#define KSZ    8
#define CCH    3
#define NOUT   1000
#define NBATCH 256
#define KDIM   192          // C*K*K
#define BETA   10.0f
#define ROWP   200          // padded LDS row stride (halves), 400B keeps b128 alignment
#define IMG    128          // D*K

// One workgroup per (w,h) location. 512 threads = 16 waves (wave32).
// Phase 1: per-row softmax fully materialized (normalized) as f16 in LDS.
// Phase 2: each wave owns 4 N-tiles (64 virtual tiles, tile 63 clamped) so W is
//          streamed from global exactly once per workgroup. Two N-tiles' B
//          fragments live in registers while looping all 16 M-tiles; A comes
//          from LDS (1x ds_load_b128 per WMMA thanks to the N-pairing).
//          12 v_wmma_f32_16x16x32_f16 per M-step; epilogue is pure stores.
__global__ __launch_bounds__(512) void localmem_wmma(
    const float* __restrict__ x, const float* __restrict__ W,
    float* __restrict__ out)
{
  __shared__ _Float16 Alds[NBATCH * ROWP];   // ~100 KB of the 320 KB LDS

  const int loc = blockIdx.x;            // w*16 + h
  const int w   = loc >> 4;
  const int h   = loc & 15;
  const int tid = threadIdx.x;

  // ---------------- Phase 1: softmax rows into LDS ----------------
  if (tid < NBATCH) {
    const float* xb = x + (size_t)tid * (CCH * IMG * IMG) + (w * KSZ) * IMG + h * KSZ;

    float mx = -3.4e38f;
    for (int c = 0; c < CCH; ++c) {
      for (int kr = 0; kr < KSZ; ++kr) {
        const float* p = xb + c * (IMG * IMG) + kr * IMG;
        v4f v0 = *(const v4f*)p;
        v4f v1 = *(const v4f*)(p + 4);
        #pragma unroll
        for (int j = 0; j < 4; ++j) {
          mx = fmaxf(mx, v0[j]);
          mx = fmaxf(mx, v1[j]);
        }
      }
    }

    float ssum = 0.0f;
    _Float16* row = &Alds[tid * ROWP];
    for (int c = 0; c < CCH; ++c) {
      for (int kr = 0; kr < KSZ; ++kr) {
        const float* p = xb + c * (IMG * IMG) + kr * IMG;   // L2-hot on 2nd pass
        v4f v0 = *(const v4f*)p;
        v4f v1 = *(const v4f*)(p + 4);
        v8h e;
        #pragma unroll
        for (int j = 0; j < 4; ++j) {
          float e0 = __expf(BETA * (v0[j] - mx));
          float e1 = __expf(BETA * (v1[j] - mx));
          ssum += e0 + e1;
          e[j]     = (_Float16)e0;
          e[4 + j] = (_Float16)e1;
        }
        *(v8h*)(row + (c * KSZ + kr) * KSZ) = e;            // ds_store_b128
      }
    }

    // Normalize in-place (v_pk_mul_f16); epilogue becomes pure stores.
    const _Float16 hs = (_Float16)(1.0f / ssum);
    #pragma unroll
    for (int q = 0; q < KDIM / 8; ++q) {
      v8h e = *(v8h*)(row + q * 8);
      #pragma unroll
      for (int j = 0; j < 8; ++j) e[j] = e[j] * hs;
      *(v8h*)(row + q * 8) = e;
    }
  }
  __syncthreads();

  // ---------------- Phase 2: per-wave WMMA GEMM ----------------
  const int lane = tid & 31;
  const int wv   = tid >> 5;          // 0..15
  const int half = lane >> 4;         // lane group (ISA VGPR layout)
  const int lo   = lane & 15;

  const float* Wl = W + (size_t)loc * (KDIM * NOUT);

  // 64 virtual N-tiles; wave wv owns {wv, wv+16, wv+32, wv+48}, processed as
  // two pairs so two B-fragment sets stay resident in registers.
  for (int p = 0; p < 2; ++p) {
    const int nt0 = wv + 32 * p;
    const int nt1 = nt0 + 16;                        // <= 63
    const int n0  = nt0 * 16 + lo;                   // always < NOUT
    const int n1  = nt1 * 16 + lo;                   // may exceed NOUT
    const float* W0 = Wl + n0;
    const float* W1 = Wl + ((n1 < NOUT) ? n1 : (NOUT - 1));  // clamp keeps EXEC uniform

    // B 32x16 f16 layout: lane column n, K = 32s + 16*half + i
    v16h b0[6], b1[6];
    #pragma unroll
    for (int s = 0; s < 6; ++s) {
      const int kb = 32 * s + 16 * half;
      #pragma unroll
      for (int i = 0; i < 16; ++i) {
        b0[s][i] = (_Float16)W0[(size_t)(kb + i) * NOUT];
        b1[s][i] = (_Float16)W1[(size_t)(kb + i) * NOUT];
      }
    }

    #pragma unroll 1
    for (int m = 0; m < 16; ++m) {
      // A 16x32 f16 layout: K = 32s + 16*half + i
      v16h a[6];
      #pragma unroll
      for (int s = 0; s < 6; ++s) {
        const _Float16* rp = &Alds[(m * 16 + lo) * ROWP + 32 * s + 16 * half];
        v8h alo = *(const v8h*)rp;          // ds_load_b128
        v8h ahi = *(const v8h*)(rp + 8);    // ds_load_b128
        a[s] = __builtin_shufflevector(alo, ahi,
                                       0, 1, 2, 3, 4, 5, 6, 7,
                                       8, 9, 10, 11, 12, 13, 14, 15);
      }

      v8f acc0 = {0.f, 0.f, 0.f, 0.f, 0.f, 0.f, 0.f, 0.f};
      v8f acc1 = {0.f, 0.f, 0.f, 0.f, 0.f, 0.f, 0.f, 0.f};
      #pragma unroll
      for (int s = 0; s < 6; ++s) {
        acc0 = __builtin_amdgcn_wmma_f32_16x16x32_f16(
                   false, a[s], false, b0[s], (short)0, acc0, false, false);
        acc1 = __builtin_amdgcn_wmma_f32_16x16x32_f16(
                   false, a[s], false, b1[s], (short)0, acc1, false, false);
      }

      // C/D layout: VGPR r -> M = r + 8*half. Common base + constant offsets
      // (r * NOUT*256*4 B < 24-bit imm) -> one address VGPR, 8 offset stores.
      {
        float* ob = out + ((size_t)(m * 16 + 8 * half) * NOUT + n0) * 256 + loc;
        #pragma unroll
        for (int r = 0; r < 8; ++r)
          ob[(size_t)r * (NOUT * 256)] = acc0[r];
      }
      if (n1 < NOUT) {   // single exec-mask toggle around the whole block
        float* ob = out + ((size_t)(m * 16 + 8 * half) * NOUT + n1) * 256 + loc;
        #pragma unroll
        for (int r = 0; r < 8; ++r)
          ob[(size_t)r * (NOUT * 256)] = acc1[r];
      }
    }
  }
}

extern "C" void kernel_launch(void* const* d_in, const int* in_sizes, int n_in,
                              void* d_out, int out_size, void* d_ws, size_t ws_size,
                              hipStream_t stream) {
  const float* x = (const float*)d_in[0];   // [256, 3, 128, 128] f32
  const float* W = (const float*)d_in[1];   // [16, 16, 192, 1000] f32
  float* out = (float*)d_out;               // [256, 1000, 16, 16] f32
  hipLaunchKernelGGL(localmem_wmma, dim3(DGRID * DGRID), dim3(512), 0, stream,
                     x, W, out);
}
